// LSTMHierarchialAttention_75247827026679
// MI455X (gfx1250) — compile-verified, matrix operand order checked by reading, not run on hardware
//
#include <hip/hip_runtime.h>
#include <hip/hip_bf16.h>
#include <math.h>

#define I_DIM 1024
#define H_DIM 1024
#define W_LEN 128
#define S_LEN 64
#define G4H   4096
#define T1    (S_LEN * W_LEN)   /* 8192 */

#define USE_ASYNC_LDS 1

typedef __bf16 bf16;
typedef bf16  v16bf __attribute__((ext_vector_type(16)));
typedef float v8f   __attribute__((ext_vector_type(8)));

union FragBF { uint4 u[2]; v16bf v; };

__device__ __forceinline__ float bflo(unsigned u) { return __builtin_bit_cast(float, u << 16); }
__device__ __forceinline__ float bfhi(unsigned u) { return __builtin_bit_cast(float, u & 0xffff0000u); }
__device__ __forceinline__ float sigf(float x) { return 1.0f / (1.0f + __expf(-x)); }

// ---------------------------------------------------------------------------
__global__ void k_zero(unsigned* ctr) {
    if (threadIdx.x < 64) ctr[threadIdx.x] = 0u;
}

// Serial softmax of the two tiny attention vectors (128 and 64 elems).
__global__ void k_softmax(const float* __restrict__ aw_in, const float* __restrict__ as_in,
                          float* __restrict__ aw, float* __restrict__ asn) {
    int t = threadIdx.x;
    if (t < 2) {
        const float* src = (t == 0) ? aw_in : as_in;
        float* dst = (t == 0) ? aw : asn;
        int n = (t == 0) ? W_LEN : S_LEN;
        float m = -3.4e38f;
        for (int i = 0; i < n; ++i) m = fmaxf(m, src[i]);
        float s = 0.f;
        for (int i = 0; i < n; ++i) s += __expf(src[i] - m);
        float inv = 1.f / s;
        for (int i = 0; i < n; ++i) dst[i] = __expf(src[i] - m) * inv;
    }
}

__global__ void k_cvt_bf16(const float* __restrict__ in, bf16* __restrict__ out, int n) {
    int i = blockIdx.x * blockDim.x + threadIdx.x;
    int stride = gridDim.x * blockDim.x;
    for (; i < n; i += stride) out[i] = (bf16)in[i];
}

// ---------------------------------------------------------------------------
// C[M x N] = A[M x K] * B[N x K]^T + (bias0 + bias1)[N]
// A, B bf16; C fp32. Block: 256 thr (8 waves). Block tile M=128 x N=64,
// wave tile 16x64: 4 x v_wmma_f32_16x16x32_bf16 per K-step sharing one A frag.
// B tile (64 n-rows x 32 k = 4 KB) staged via async DMA to LDS (ASYNCcnt).
__global__ void k_gemm_bf16(const bf16* __restrict__ A, const bf16* __restrict__ B,
                            const float* __restrict__ bias0, const float* __restrict__ bias1,
                            float* __restrict__ C, int M, int N, int K) {
    __shared__ uint4 lds_b4[256];   // 64(n) x 32(k) bf16 = 4 KB

    const int tid   = threadIdx.x;
    const int lane  = tid & 31;
    const int wid   = tid >> 5;
    const int n0    = blockIdx.x * 64;
    const int m0    = blockIdx.y * 128 + wid * 16;
    const int l15   = lane & 15;
    const int kbase = (lane >> 4) * 8;    // K sub-block offset per ISA A/B layout
    const bool active = (m0 < M);         // wave-uniform; inactive waves still stage B

#if USE_ASYNC_LDS
    // wave-relative LDS byte offset of this thread's staging slot
    typedef __attribute__((address_space(3))) char lds_char;
    const unsigned ldsoff =
        (unsigned)(unsigned long long)(lds_char*)lds_b4 + (unsigned)tid * 16u;
#endif

    const bf16* arow = A + (size_t)(m0 + l15) * K;
    const bf16* bsrc = B + (size_t)(n0 + (tid >> 2)) * K + (tid & 3) * 8;

    v8f acc[4];
    #pragma unroll
    for (int nt = 0; nt < 4; ++nt) acc[nt] = (v8f){0.f,0.f,0.f,0.f,0.f,0.f,0.f,0.f};

    for (int k0 = 0; k0 < K; k0 += 32) {
        __syncthreads();
        // ---- stage B tile: each thread moves 16 B (8 bf16) ----
#if USE_ASYNC_LDS
        asm volatile("global_load_async_to_lds_b128 %0, %1, off"
                     :: "v"(ldsoff), "v"(bsrc + k0)
                     : "memory");
        asm volatile("s_wait_asynccnt 0x0" ::: "memory");
#else
        lds_b4[tid] = *(const uint4*)(bsrc + k0);
#endif
        if (k0 + 32 < K) {
            __builtin_prefetch(arow + k0 + 32 + kbase, 0, 3);   // global_prefetch
        }
        __syncthreads();

        if (active) {
            FragBF fa, fb[4];
            // A 16x32: elems 0-7 at K = kbase.., elems 8-15 at K = 16+kbase..
            fa.u[0] = *(const uint4*)(arow + k0 + kbase);
            fa.u[1] = *(const uint4*)(arow + k0 + 16 + kbase);
            const bf16* lb = (const bf16*)lds_b4;
            #pragma unroll
            for (int nt = 0; nt < 4; ++nt) {
                const bf16* lrow = lb + (nt * 16 + l15) * 32;
                fb[nt].u[0] = *(const uint4*)(lrow + kbase);
                fb[nt].u[1] = *(const uint4*)(lrow + 16 + kbase);
            }
            #pragma unroll
            for (int nt = 0; nt < 4; ++nt) {
                acc[nt] = __builtin_amdgcn_wmma_f32_16x16x32_bf16(
                    false, fa.v, false, fb[nt].v, (short)0, acc[nt], false, false);
            }
        }
    }

    if (active) {
        // C/D layout: VGPR r holds M = m0 + (lane>=16 ? 8 : 0) + r, N = n + (lane&15)
        int mrow = m0 + ((lane >> 4) << 3);
        #pragma unroll
        for (int nt = 0; nt < 4; ++nt) {
            int n = n0 + nt * 16 + l15;
            float bias = bias0[n] + bias1[n];
            #pragma unroll
            for (int r = 0; r < 8; ++r) {
                C[(size_t)(mrow + r) * N + n] = acc[nt][r] + bias;
            }
        }
    }
}

// ---------------------------------------------------------------------------
// Sequential LSTM recurrence with fused attention pooling.
// Grid: 16 blocks x 256 threads (cooperative, atomic-counter grid barrier).
// Block b owns hidden slice [b*64, b*64+64) and the matching 4x64 gate rows.
// G[t] already contains x@Wih^T + bih + bhh.
__global__ void k_lstm_recur(const float* __restrict__ G,    // [T x 4096]
                             const bf16*  __restrict__ U,    // [4096 x 1024] Whh (bf16)
                             const float* __restrict__ h0,
                             const float* __restrict__ c0,
                             const float* __restrict__ pw,   // pooling weights [wlen]
                             float* __restrict__ pool_out,   // [T/wlen x 1024]
                             float* __restrict__ Hbuf,       // [2 x 1024] double-buffered h
                             unsigned* __restrict__ ctr,
                             int T, int wlen) {
    __shared__ float lds_h[H_DIM];
    __shared__ float lds_g[256];
    __shared__ float lds_c[64];

    const int tid = threadIdx.x;
    const int b   = blockIdx.x;
    const int q   = tid >> 6;                 // gate: 0=i 1=f 2=g 3=o
    const int jj  = tid & 63;
    const int row = q * H_DIM + b * 64 + jj;  // row of Whh / G
    const unsigned NB = gridDim.x;

    if (tid < 64) {
        int hidx = b * 64 + tid;
        Hbuf[hidx] = h0[hidx];                // buffer 0 <- h_init
        lds_c[tid] = c0[hidx];
        __threadfence();
    }
    float encacc = 0.f;

    __syncthreads();
    if (tid == 0) {
        __hip_atomic_fetch_add(ctr, 1u, __ATOMIC_RELEASE, __HIP_MEMORY_SCOPE_AGENT);
        while (__hip_atomic_load(ctr, __ATOMIC_ACQUIRE, __HIP_MEMORY_SCOPE_AGENT) < NB)
            __builtin_amdgcn_s_sleep(1);
    }
    __syncthreads();

    const uint4* urow = (const uint4*)(U + (size_t)row * H_DIM);  // 128 x 16B

    for (int t = 0; t < T; ++t) {
        const float* Hr = Hbuf + (t & 1) * H_DIM;
        float*       Hw = Hbuf + ((t + 1) & 1) * H_DIM;

        for (int i = tid; i < H_DIM; i += 256) lds_h[i] = Hr[i];
        __syncthreads();

        float acc = G[(size_t)t * G4H + row];
        const float4* lh = (const float4*)lds_h;
        #pragma unroll 4
        for (int k = 0; k < H_DIM / 8; ++k) {
            uint4 u = urow[k];
            float4 ha = lh[2 * k], hb = lh[2 * k + 1];
            acc += bflo(u.x) * ha.x + bfhi(u.x) * ha.y
                 + bflo(u.y) * ha.z + bfhi(u.y) * ha.w
                 + bflo(u.z) * hb.x + bfhi(u.z) * hb.y
                 + bflo(u.w) * hb.z + bfhi(u.w) * hb.w;
        }
        lds_g[tid] = acc;
        __syncthreads();

        if (tid < 64) {
            float gi = lds_g[tid];
            float gf = lds_g[64 + tid];
            float gg = lds_g[128 + tid];
            float go = lds_g[192 + tid];
            float c  = sigf(gf) * lds_c[tid] + sigf(gi) * tanhf(gg);
            float h  = sigf(go) * tanhf(c);
            lds_c[tid] = c;
            Hw[b * 64 + tid] = h;
            int w = t - (t / wlen) * wlen;
            encacc += pw[w] * h;
            if (w == wlen - 1) {
                pool_out[(size_t)(t / wlen) * H_DIM + b * 64 + tid] = encacc;
                encacc = 0.f;
            }
            __threadfence();
        }

        __syncthreads();
        if (tid == 0) {
            __hip_atomic_fetch_add(ctr, 1u, __ATOMIC_RELEASE, __HIP_MEMORY_SCOPE_AGENT);
            unsigned target = (unsigned)(t + 2) * NB;
            while (__hip_atomic_load(ctr, __ATOMIC_ACQUIRE, __HIP_MEMORY_SCOPE_AGENT) < target)
                __builtin_amdgcn_s_sleep(1);
        }
        __syncthreads();
    }
}

// ---------------------------------------------------------------------------
__global__ void k_final(const float* __restrict__ rowv, const float* __restrict__ Wf,
                        const float* __restrict__ bfin, float* __restrict__ out) {
    __shared__ float red[256];
    int tid = threadIdx.x;
    float s = 0.f;
    for (int i = tid; i < H_DIM; i += 256) s += rowv[i] * Wf[i];
    red[tid] = s;
    __syncthreads();
    for (int off = 128; off > 0; off >>= 1) {
        if (tid < off) red[tid] += red[tid + off];
        __syncthreads();
    }
    if (tid == 0) out[0] = 1.f / (1.f + __expf(-(red[0] + bfin[0])));
}

// ---------------------------------------------------------------------------
extern "C" void kernel_launch(void* const* d_in, const int* in_sizes, int n_in,
                              void* d_out, int out_size, void* d_ws, size_t ws_size,
                              hipStream_t stream) {
    const float* X    = (const float*)d_in[0];
    const float* h1   = (const float*)d_in[1];
    const float* c1   = (const float*)d_in[2];
    const float* h2   = (const float*)d_in[3];
    const float* c2   = (const float*)d_in[4];
    const float* Wih1 = (const float*)d_in[5];
    const float* Whh1 = (const float*)d_in[6];
    const float* bih1 = (const float*)d_in[7];
    const float* bhh1 = (const float*)d_in[8];
    const float* Wih2 = (const float*)d_in[9];
    const float* Whh2 = (const float*)d_in[10];
    const float* bih2 = (const float*)d_in[11];
    const float* bhh2 = (const float*)d_in[12];
    const float* attw = (const float*)d_in[13];
    const float* atts = (const float*)d_in[14];
    const float* Wf   = (const float*)d_in[15];
    const float* bfin = (const float*)d_in[16];
    float* out = (float*)d_out;
    (void)in_sizes; (void)n_in; (void)out_size; (void)ws_size;

    char* ws = (char*)d_ws;
    size_t off = 0;
    auto alloc = [&](size_t bytes) {
        size_t r = off;
        off += (bytes + 255) & ~(size_t)255;
        return r;
    };
    unsigned* ctr  = (unsigned*)(ws + alloc(256));          // [0]=layer1, [32]=layer2
    float* aw      = (float*)(ws + alloc(W_LEN * 4));
    float* asn     = (float*)(ws + alloc(S_LEN * 4));
    float* Hb1     = (float*)(ws + alloc(2 * H_DIM * 4));
    float* Hb2     = (float*)(ws + alloc(2 * H_DIM * 4));
    float* rowv    = (float*)(ws + alloc(H_DIM * 4));
    float* encs    = (float*)(ws + alloc((size_t)S_LEN * H_DIM * 4));
    bf16*  encsb   = (bf16*)(ws + alloc((size_t)S_LEN * H_DIM * 2));
    bf16*  Xbf     = (bf16*)(ws + alloc((size_t)T1 * I_DIM * 2));
    bf16*  W1b     = (bf16*)(ws + alloc((size_t)G4H * I_DIM * 2));
    bf16*  U1b     = (bf16*)(ws + alloc((size_t)G4H * H_DIM * 2));
    bf16*  W2b     = (bf16*)(ws + alloc((size_t)G4H * H_DIM * 2));
    bf16*  U2b     = (bf16*)(ws + alloc((size_t)G4H * H_DIM * 2));
    float* G1      = (float*)(ws + alloc((size_t)T1 * G4H * 4));
    float* G2      = (float*)(ws + alloc((size_t)S_LEN * G4H * 4));

    k_zero<<<1, 64, 0, stream>>>(ctr);
    k_softmax<<<1, 2, 0, stream>>>(attw, atts, aw, asn);

    k_cvt_bf16<<<2048, 256, 0, stream>>>(X,    Xbf, T1 * I_DIM);
    k_cvt_bf16<<<512,  256, 0, stream>>>(Wih1, W1b, G4H * I_DIM);
    k_cvt_bf16<<<512,  256, 0, stream>>>(Whh1, U1b, G4H * H_DIM);
    k_cvt_bf16<<<512,  256, 0, stream>>>(Wih2, W2b, G4H * H_DIM);
    k_cvt_bf16<<<512,  256, 0, stream>>>(Whh2, U2b, G4H * H_DIM);

    // G1 = X @ Wih1^T + bih1 + bhh1   (M=8192, N=4096, K=1024)
    k_gemm_bf16<<<dim3(G4H / 64, T1 / 128), 256, 0, stream>>>(
        Xbf, W1b, bih1, bhh1, G1, T1, G4H, I_DIM);

    // word-level recurrence + attention pooling into encs
    k_lstm_recur<<<16, 256, 0, stream>>>(G1, U1b, h1, c1, aw, encs, Hb1,
                                         ctr + 0, T1, W_LEN);

    k_cvt_bf16<<<64, 256, 0, stream>>>(encs, encsb, S_LEN * H_DIM);

    // G2 = encs @ Wih2^T + bih2 + bhh2   (M=64, N=4096, K=1024)
    k_gemm_bf16<<<dim3(G4H / 64, 1), 256, 0, stream>>>(
        encsb, W2b, bih2, bhh2, G2, S_LEN, G4H, H_DIM);

    // sentence-level recurrence + attention pooling into rowv
    k_lstm_recur<<<16, 256, 0, stream>>>(G2, U2b, h2, c2, asn, rowv, Hb2,
                                         ctr + 32, S_LEN, S_LEN);

    k_final<<<1, 256, 0, stream>>>(rowv, Wf, bfin, out);
}
